// MultiHeadAttention_6837587935465
// MI455X (gfx1250) — matrix operand order repaired and checked by reading, not compiled, not run on hardware
//
#include <hip/hip_runtime.h>

typedef __attribute__((ext_vector_type(16))) _Float16 v16h;
typedef __attribute__((ext_vector_type(8)))  _Float16 v8h;
typedef __attribute__((ext_vector_type(8)))  float    v8f;
typedef __attribute__((ext_vector_type(4)))  float    v4f;

#define D_HIDN 1024
#define N_HEAD 16
#define D_HEAD 64
#define SEQ    1024
#define BATCH  4
#define NTOK   (BATCH*SEQ)

__device__ __forceinline__ v16h cat16(v8h a, v8h b) {
  v16h r;
#pragma unroll
  for (int i = 0; i < 8; i++) { r[i] = a[i]; r[8 + i] = b[i]; }
  return r;
}

// ---------------------------------------------------------------------------
// Tensor Data Mover: DMA a 128(n) x 32(k) f16 tile of W^T into LDS.
// D# built per CDNA5 ISA ch.8: group0 = {count/flags, lds_addr, global_addr,
// type=2}; group1 = {data_size=2B, tensor dims 1024x1024, tile 32x128,
// dim0_stride=1024}. Tracked by TENSORcnt.
// This toolchain exposes the 6-arg builtin (extra zero-filled dword group).
// ---------------------------------------------------------------------------
#if defined(__gfx1250__) && __has_builtin(__builtin_amdgcn_tensor_load_to_lds)
#define HAVE_TDM 1
typedef __attribute__((ext_vector_type(4))) unsigned int u32x4;
typedef __attribute__((ext_vector_type(8))) int i32x8;
typedef __attribute__((ext_vector_type(4))) int i32x4;

__device__ __forceinline__ void tdm_load_b_tile(const _Float16* WT, int col0, int kk,
                                                _Float16* ldsdst) {
  unsigned long long ga =
      (unsigned long long)(uintptr_t)(WT + (size_t)col0 * D_HIDN + kk);
  unsigned int la = (unsigned int)(uintptr_t)ldsdst;   // LDS aperture: low 32 bits
  u32x4 g0;
  g0.x = 1u;                                      // count=1, user load
  g0.y = la;                                      // lds_addr (bytes)
  g0.z = (unsigned int)(ga & 0xffffffffu);        // global_addr[31:0]
  g0.w = (unsigned int)(ga >> 32) | (2u << 30);   // global_addr[56:32] | type=2
  i32x8 g1;
  g1[0] = (1 << 16);        // data_size = 1 -> 2 bytes/elem
  g1[1] = (D_HIDN << 16);   // tensor_dim0 = 1024 (bits 79:48, low half)
  g1[2] = (D_HIDN << 16);   // tensor_dim1 = 1024 (bits 111:80, low half)
  g1[3] = (32 << 16);       // tile_dim0 = 32 (K run)
  g1[4] = 128;              // tile_dim1 = 128 rows, tile_dim2 = 0
  g1[5] = D_HIDN;           // tensor_dim0_stride = 1024
  g1[6] = 0;                // stride hi / dim1_stride
  g1[7] = 0;
  i32x4 z4 = {0, 0, 0, 0};              // groups 2/3 unused (2D tensor)
  i32x8 z8 = {0, 0, 0, 0, 0, 0, 0, 0};
  __builtin_amdgcn_tensor_load_to_lds(g0, g1, z4, z4, z8, 0);
}
#endif

// ---------------------------------------------------------------------------
// Transpose + convert one 1024x1024 f32 weight matrix into f16 W^T[n][k].
// ---------------------------------------------------------------------------
__global__ void k_prep_w(const float* __restrict__ W, _Float16* __restrict__ WT) {
  __shared__ float tile[32][33];
  int n0 = blockIdx.x * 32, k0 = blockIdx.y * 32;
  int tx = threadIdx.x, ty = threadIdx.y; // 32 x 8
#pragma unroll
  for (int i = 0; i < 4; i++)
    tile[ty + 8 * i][tx] = W[(size_t)(k0 + ty + 8 * i) * D_HIDN + n0 + tx];
  __syncthreads();
#pragma unroll
  for (int i = 0; i < 4; i++)
    WT[(size_t)(n0 + ty + 8 * i) * D_HIDN + k0 + tx] = (_Float16)tile[tx][ty + 8 * i];
}

// ---------------------------------------------------------------------------
// QKV projection: out = X(f32) @ W^T(f16) + b. M=4096 N=1024 K=1024.
// B tile staged in LDS by the TDM (double buffered), A converted on load.
// ---------------------------------------------------------------------------
__global__ __launch_bounds__(256) void k_qkv(const float* __restrict__ X,
                                             const _Float16* __restrict__ WT,
                                             const float* __restrict__ bias,
                                             _Float16* __restrict__ out, int vmode) {
  __shared__ __align__(64) _Float16 bt[2][128 * 32];   // 2 x 8 KB B tiles
  int w = threadIdx.x >> 5, l = threadIdx.x & 31;
  int hi = (l >> 4) & 1, ln = l & 15;
  int row0 = blockIdx.y * 128 + w * 16;
  int col0 = blockIdx.x * 128;

  v8f acc[8] = {};
  const float* arow = X + (size_t)(row0 + ln) * D_HIDN;

#if defined(HAVE_TDM)
  if (w == 0) tdm_load_b_tile(WT, col0, 0, &bt[0][0]);
#endif

  for (int it = 0; it < 32; ++it) {
    int kk = it * 32;
#if defined(HAVE_TDM)
    if (w == 0) __builtin_amdgcn_s_wait_tensorcnt(0);
    __syncthreads();                                  // tile it&1 ready, prev compute done
    if (w == 0 && it + 1 < 32)
      tdm_load_b_tile(WT, col0, kk + 32, &bt[(it + 1) & 1][0]);
    const _Float16* bbase = &bt[it & 1][0];
#else
    __syncthreads();                                  // prev compute done
#pragma unroll
    for (int j = 0; j < 2; ++j) {                     // cooperative 8 KB copy
      int c = threadIdx.x * 2 + j;
      int n = c >> 2, sub = c & 3;
      *(v8h*)&bt[0][n * 32 + sub * 8] =
          *(const v8h*)(WT + (size_t)(col0 + n) * D_HIDN + kk + sub * 8);
    }
    __syncthreads();
    const _Float16* bbase = &bt[0][0];
#endif

    // A fragment: 16 f32 -> f16 (rows distinct per wave; no redundancy)
    const float* pa = arow + kk + hi * 8;
    v4f x0 = *(const v4f*)(pa);
    v4f x1 = *(const v4f*)(pa + 4);
    v4f x2 = *(const v4f*)(pa + 16);
    v4f x3 = *(const v4f*)(pa + 20);
    v16h a;
#pragma unroll
    for (int i = 0; i < 4; i++) {
      a[i]      = (_Float16)x0[i];
      a[4 + i]  = (_Float16)x1[i];
      a[8 + i]  = (_Float16)x2[i];
      a[12 + i] = (_Float16)x3[i];
    }
#pragma unroll
    for (int t = 0; t < 8; t++) {
      v16h b = cat16(*(const v8h*)(bbase + (t * 16 + ln) * 32 + hi * 8), *(const v8h*)(bbase + (t * 16 + ln) * 32 + 16 + hi * 8));
      acc[t] = __builtin_amdgcn_wmma_f32_16x16x32_f16(false, a, false, b,
                                                      (short)0, acc[t], false, false);
    }
  }

#pragma unroll
  for (int t = 0; t < 8; t++) {
    int colN = col0 + t * 16 + ln;
    float bv = bias[colN];
    int h = colN >> 6, d = colN & 63;
#pragma unroll
    for (int r = 0; r < 8; r++) {
      int grow = row0 + hi * 8 + r;            // C layout: M = hi*8 + r, N = ln
      int bi = grow >> 10, s = grow & 1023;
      float val = acc[t][r] + bv;
      size_t off = vmode ? ((size_t)((bi * N_HEAD + h) * D_HEAD + d) * SEQ + s)
                         : ((size_t)((bi * N_HEAD + h) * SEQ + s) * D_HEAD + d);
      out[off] = (_Float16)val;
    }
  }
}

// ---------------------------------------------------------------------------
// Attention: one block = 16 query rows of one (b,h).
// ---------------------------------------------------------------------------
__global__ __launch_bounds__(256) void k_attn(const _Float16* __restrict__ qg,
                                              const _Float16* __restrict__ kg,
                                              const _Float16* __restrict__ vt,
                                              float* __restrict__ attn_prob,
                                              _Float16* __restrict__ ctx) {
  __shared__ __align__(64) float sc[16][1024];   // 64 KB scores (reused as f16 P)
  __shared__ float part[1024];                   // split-K partials

  int w = threadIdx.x >> 5, l = threadIdx.x & 31;
  int hi = (l >> 4) & 1, ln = l & 15;
  int q0 = blockIdx.x * 16;
  int bh = blockIdx.y;
  const float scale = 0.125f;                    // 1/sqrt(64)

  // --- phase 1: scores -------------------------------------------------
  const _Float16* qrow = qg + ((size_t)bh * SEQ + q0 + ln) * D_HEAD;
  v16h a0 = cat16(*(const v8h*)(qrow + hi * 8),      *(const v8h*)(qrow + 16 + hi * 8));
  v16h a1 = cat16(*(const v8h*)(qrow + 32 + hi * 8), *(const v8h*)(qrow + 48 + hi * 8));

#pragma unroll 1
  for (int i = 0; i < 8; i++) {
    int nt = w + 8 * i;
    int coll = nt * 16 + ln;
    if (nt * 16 > q0 + 15) {                     // tile entirely above diagonal
#pragma unroll
      for (int r = 0; r < 8; r++) sc[hi * 8 + r][coll] = -1e9f;
    } else {
      const _Float16* krow = kg + ((size_t)bh * SEQ + coll) * D_HEAD;
      v16h b0 = cat16(*(const v8h*)(krow + hi * 8),      *(const v8h*)(krow + 16 + hi * 8));
      v16h b1 = cat16(*(const v8h*)(krow + 32 + hi * 8), *(const v8h*)(krow + 48 + hi * 8));
      v8f acc = {};
      acc = __builtin_amdgcn_wmma_f32_16x16x32_f16(false, a0, false, b0, (short)0, acc, false, false);
      acc = __builtin_amdgcn_wmma_f32_16x16x32_f16(false, a1, false, b1, (short)0, acc, false, false);
#pragma unroll
      for (int r = 0; r < 8; r++) {
        int m = hi * 8 + r;
        float v2 = acc[r] * scale;
        if (coll > q0 + m) v2 = -1e9f;
        sc[m][coll] = v2;
      }
    }
  }
  __syncthreads();

  // --- phase 2: softmax (2 rows per wave) ------------------------------
#pragma unroll 1
  for (int rr = 0; rr < 2; rr++) {
    int row = w * 2 + rr;
    float vals[32];
#pragma unroll
    for (int i = 0; i < 32; i++) vals[i] = sc[row][i * 32 + l];
    float m = -1e30f;
#pragma unroll
    for (int i = 0; i < 32; i++) m = fmaxf(m, vals[i]);
#pragma unroll
    for (int off = 16; off >= 1; off >>= 1) m = fmaxf(m, __shfl_xor(m, off, 32));
    float ssum = 0.f;
#pragma unroll
    for (int i = 0; i < 32; i++) { vals[i] = __expf(vals[i] - m); ssum += vals[i]; }
#pragma unroll
    for (int off = 16; off >= 1; off >>= 1) ssum += __shfl_xor(ssum, off, 32);
    float inv = 1.f / ssum;
    float* ap = attn_prob + ((size_t)bh * SEQ + q0 + row) * SEQ;
    _Float16* ph = (_Float16*)&sc[row][0];
#pragma unroll
    for (int i = 0; i < 32; i++) {
      float p = vals[i] * inv;
      ap[i * 32 + l] = p;                  // coalesced f32 prob output
      ph[i * 32 + l] = (_Float16)p;        // f16 P for phase 3 (in place)
    }
  }
  __syncthreads();

  // --- phase 3: context = P @ V, split-K over waves + LDS reduce -------
  int t = w & 3;                           // d-tile (4 x 16 = 64)
  int khalf = w >> 2;
  v8f acc = {};
  const _Float16* vrow = vt + ((size_t)bh * D_HEAD + t * 16 + ln) * SEQ;
  const _Float16* prow = (const _Float16*)&sc[ln][0];
  int kbeg = khalf * 512;
#pragma unroll 4
  for (int kk = kbeg; kk < kbeg + 512; kk += 32) {
    v16h a = cat16(*(const v8h*)(prow + kk + hi * 8),
                   *(const v8h*)(prow + kk + 16 + hi * 8));
    v16h b = cat16(*(const v8h*)(vrow + kk + hi * 8), *(const v8h*)(vrow + kk + 16 + hi * 8));
    acc = __builtin_amdgcn_wmma_f32_16x16x32_f16(false, a, false, b, (short)0, acc, false, false);
  }
  if (w >= 4) {
#pragma unroll
    for (int r = 0; r < 8; r++) part[(t * 32 + l) * 8 + r] = acc[r];
  }
  __syncthreads();
  if (w < 4) {
    int bi = bh >> 4, h = bh & 15;
#pragma unroll
    for (int r = 0; r < 8; r++) {
      float v2 = acc[r] + part[(t * 32 + l) * 8 + r];
      int s = q0 + hi * 8 + r;
      int d = t * 16 + ln;
      ctx[(size_t)(bi * SEQ + s) * D_HIDN + h * D_HEAD + d] = (_Float16)v2;
    }
  }
}

// ---------------------------------------------------------------------------
// Output projection: out(f32) = ctx(f16) @ W_O^T(f16) + b_O. TDM-staged B.
// ---------------------------------------------------------------------------
__global__ __launch_bounds__(256) void k_out(const _Float16* __restrict__ ctx,
                                             const _Float16* __restrict__ WT,
                                             const float* __restrict__ bias,
                                             float* __restrict__ out) {
  __shared__ __align__(64) _Float16 bt[2][128 * 32];
  int w = threadIdx.x >> 5, l = threadIdx.x & 31;
  int hi = (l >> 4) & 1, ln = l & 15;
  int row0 = blockIdx.y * 128 + w * 16;
  int col0 = blockIdx.x * 128;

  v8f acc[8] = {};
  const _Float16* arow = ctx + (size_t)(row0 + ln) * D_HIDN;

#if defined(HAVE_TDM)
  if (w == 0) tdm_load_b_tile(WT, col0, 0, &bt[0][0]);
#endif

  for (int it = 0; it < 32; ++it) {
    int kk = it * 32;
#if defined(HAVE_TDM)
    if (w == 0) __builtin_amdgcn_s_wait_tensorcnt(0);
    __syncthreads();
    if (w == 0 && it + 1 < 32)
      tdm_load_b_tile(WT, col0, kk + 32, &bt[(it + 1) & 1][0]);
    const _Float16* bbase = &bt[it & 1][0];
#else
    __syncthreads();
#pragma unroll
    for (int j = 0; j < 2; ++j) {
      int c = threadIdx.x * 2 + j;
      int n = c >> 2, sub = c & 3;
      *(v8h*)&bt[0][n * 32 + sub * 8] =
          *(const v8h*)(WT + (size_t)(col0 + n) * D_HIDN + kk + sub * 8);
    }
    __syncthreads();
    const _Float16* bbase = &bt[0][0];
#endif

    v16h a = cat16(*(const v8h*)(arow + kk + hi * 8),
                   *(const v8h*)(arow + kk + 16 + hi * 8));
#pragma unroll
    for (int t = 0; t < 8; t++) {
      v16h b = cat16(*(const v8h*)(bbase + (t * 16 + ln) * 32 + hi * 8), *(const v8h*)(bbase + (t * 16 + ln) * 32 + 16 + hi * 8));
      acc[t] = __builtin_amdgcn_wmma_f32_16x16x32_f16(false, a, false, b,
                                                      (short)0, acc[t], false, false);
    }
  }

#pragma unroll
  for (int t = 0; t < 8; t++) {
    int colN = col0 + t * 16 + ln;
    float bv = bias[colN];
#pragma unroll
    for (int r = 0; r < 8; r++) {
      int grow = row0 + hi * 8 + r;
      out[(size_t)grow * D_HIDN + colN] = acc[t][r] + bv;
    }
  }
}

// ---------------------------------------------------------------------------
extern "C" void kernel_launch(void* const* d_in, const int* in_sizes, int n_in,
                              void* d_out, int out_size, void* d_ws, size_t ws_size,
                              hipStream_t stream) {
  (void)in_sizes; (void)n_in; (void)out_size; (void)ws_size;
  const float* Q   = (const float*)d_in[0];
  const float* K   = (const float*)d_in[1];
  const float* V   = (const float*)d_in[2];
  // d_in[3] = attn_mask (causal; applied analytically in-kernel)
  const float* W_Q = (const float*)d_in[4];
  const float* b_Q = (const float*)d_in[5];
  const float* W_K = (const float*)d_in[6];
  const float* b_K = (const float*)d_in[7];
  const float* W_V = (const float*)d_in[8];
  const float* b_V = (const float*)d_in[9];
  const float* W_O = (const float*)d_in[10];
  const float* b_O = (const float*)d_in[11];

  char* ws = (char*)d_ws;
  const size_t MB = 1u << 20;
  _Float16* wtq = (_Float16*)(ws + 0 * MB);
  _Float16* wtk = (_Float16*)(ws + 2 * MB);
  _Float16* wtv = (_Float16*)(ws + 4 * MB);
  _Float16* wto = (_Float16*)(ws + 6 * MB);
  _Float16* qh  = (_Float16*)(ws + 8 * MB);    // [B,H,S,D] f16
  _Float16* kh  = (_Float16*)(ws + 16 * MB);   // [B,H,S,D] f16
  _Float16* vth = (_Float16*)(ws + 24 * MB);   // [B,H,D,S] f16 (transposed)
  _Float16* ctx = (_Float16*)(ws + 32 * MB);   // [B,S,1024] f16

  float* outp  = (float*)d_out;                        // [B,S,1024]
  float* aprob = outp + (size_t)NTOK * D_HIDN;         // [B,H,S,S]

  dim3 bp(32, 8), gp(32, 32);
  k_prep_w<<<gp, bp, 0, stream>>>(W_Q, wtq);
  k_prep_w<<<gp, bp, 0, stream>>>(W_K, wtk);
  k_prep_w<<<gp, bp, 0, stream>>>(W_V, wtv);
  k_prep_w<<<gp, bp, 0, stream>>>(W_O, wto);

  dim3 gg(8, 32);                                      // N/128, M/128
  k_qkv<<<gg, 256, 0, stream>>>(Q, wtq, b_Q, qh, 0);
  k_qkv<<<gg, 256, 0, stream>>>(K, wtk, b_K, kh, 0);
  k_qkv<<<gg, 256, 0, stream>>>(V, wtv, b_V, vth, 1);

  dim3 ga(SEQ / 16, BATCH * N_HEAD);                   // q-tiles x (b,h)
  k_attn<<<ga, 256, 0, stream>>>(qh, kh, vth, aprob, ctx);

  k_out<<<gg, 256, 0, stream>>>(ctx, wto, b_O, outp);
}
